// InducedSelfAttentionBlock_43241730736533
// MI455X (gfx1250) — matrix operand order repaired and checked by reading, not compiled, not run on hardware
//
#include <hip/hip_runtime.h>
#include <math.h>

// ---------------------------------------------------------------------------
// ISAB (Induced Set Attention Block) for MI455X / gfx1250, wave32 + WMMA f16.
// B=8, N=8192, D=256, H=8, HD=32, M_IND=16, DFF=1024.
// ---------------------------------------------------------------------------

typedef _Float16 h16;
typedef __attribute__((ext_vector_type(16))) _Float16 v16h;
typedef __attribute__((ext_vector_type(8)))  _Float16 v8h;
typedef __attribute__((ext_vector_type(8)))  float    v8f;

#define BATCH   8
#define SEQ     8192
#define DMODEL  256
#define NHEAD   8
#define HEADD   32
#define MIND    16
#define DFF     1024
#define BN      (BATCH * SEQ)      // 65536 rows
#define BM      (BATCH * MIND)     // 128 rows
#define LN_EPS  1e-5f
#define SM_SCALE 0.17677669529663687f   // 1/sqrt(32)

// ------------------------- WMMA fragment helpers ---------------------------
// A (16xK=32, f16): lane row = lane&15; lanes<16 hold K 0..15,
// lanes>=16 hold K 16..31 (same k order as B fragments).
__device__ __forceinline__ v16h load_a_frag(const h16* __restrict__ A,
                                            int lda, int m0, int k0, int lane) {
  int row = lane & 15;
  int kb  = (lane >> 4) * 16;
  const h16* p = A + (size_t)(m0 + row) * lda + k0 + kb;
  v8h lo = *(const v8h*)(p);        // K = kb + 0..7
  v8h hi = *(const v8h*)(p + 8);    // K = kb + 8..15
  v16h a;
#pragma unroll
  for (int i = 0; i < 8; ++i) { a[i] = lo[i]; a[i + 8] = hi[i]; }
  return a;
}

// B (K=32 x 16, f16) from row-major W[N][K] (computes A·Wᵀ):
// lane col = lane&15; lanes<16 hold K 0..15, lanes>=16 hold K 16..31.
__device__ __forceinline__ v16h load_b_frag(const h16* __restrict__ W,
                                            int ldw, int n0, int k0, int lane) {
  int col = lane & 15;
  int kb  = (lane >> 4) * 16;
  return *(const v16h*)(W + (size_t)(n0 + col) * ldw + k0 + kb);
}

__device__ __forceinline__ v8f wmma_f16(v16h a, v16h b, v8f c) {
  return __builtin_amdgcn_wmma_f32_16x16x32_f16(false, a, false, b,
                                                (short)0, c, false, false);
}

__device__ __forceinline__ float gelu_exact(float x) {
  return 0.5f * x * (1.0f + erff(x * 0.70710678118654752f));
}

// ------------------------------ LayerNorm ----------------------------------
// One wave per 256-wide row; f32 in, f16 out.
__global__ __launch_bounds__(256) void layernorm_kernel(
    const float* __restrict__ x, const float* __restrict__ w,
    const float* __restrict__ b, h16* __restrict__ out, int rows) {
  int lane = threadIdx.x & 31;
  int row  = blockIdx.x * 8 + (threadIdx.x >> 5);
  if (row >= rows) return;
  const float* xr = x + (size_t)row * DMODEL;
  float v[8], s = 0.f, sq = 0.f;
#pragma unroll
  for (int j = 0; j < 8; ++j) {
    float t = xr[lane + 32 * j];
    v[j] = t; s += t; sq += t * t;
  }
#pragma unroll
  for (int m = 16; m >= 1; m >>= 1) {
    s  += __shfl_xor(s, m, 32);
    sq += __shfl_xor(sq, m, 32);
  }
  float mu   = s * (1.f / DMODEL);
  float var  = sq * (1.f / DMODEL) - mu * mu;
  float rstd = rsqrtf(var + LN_EPS);
  h16* orow = out + (size_t)row * DMODEL;
#pragma unroll
  for (int j = 0; j < 8; ++j) {
    int d = lane + 32 * j;
    orow[d] = (h16)((v[j] - mu) * rstd * w[d] + b[d]);
  }
}

// --------------------------- big WMMA GEMM ---------------------------------
// C[M x Nc] = act(A[M x K](f16) · W[Nc x K](f16)^T + bias) (+res[M x Nc](f32))
// 8 waves per block; each wave owns a 32x32 register tile (2x2 WMMA tiles,
// 4 wmma per 4 fragment loads per K-step). Block tile = 128 x 64.
// OUT16T stores f16 transposed as out16t[col][row] (for per-head V^T), using
// one packed v8h store per accumulator (8 consecutive rows per lane).
template <bool GELU, bool HAS_RES, bool OUT32, bool OUT16, bool OUT16T>
__global__ __launch_bounds__(256) void gemm_wmma_kernel(
    const h16* __restrict__ A, const h16* __restrict__ W,
    const float* __restrict__ bias, const float* __restrict__ resid,
    float* __restrict__ out32, h16* __restrict__ out16, h16* __restrict__ out16t,
    int M, int Nc, int K) {
  int lane = threadIdx.x & 31;
  int wave = threadIdx.x >> 5;
  int m0 = blockIdx.x * 128 + (wave & 3) * 32;
  int n0 = blockIdx.y * 64  + (wave >> 2) * 32;
  v8f acc00 = {}, acc01 = {}, acc10 = {}, acc11 = {};
  for (int k0 = 0; k0 < K; k0 += 32) {
    if (k0 + 32 < K) {   // pull next K-slice toward the WGP (global_prefetch_b8)
      __builtin_prefetch(A + (size_t)(m0 + (lane & 15)) * K + k0 + 32, 0, 1);
      __builtin_prefetch(W + (size_t)(n0 + (lane & 15)) * K + k0 + 32, 0, 1);
    }
    v16h a0 = load_a_frag(A, K, m0,      k0, lane);
    v16h a1 = load_a_frag(A, K, m0 + 16, k0, lane);
    v16h b0 = load_b_frag(W, K, n0,      k0, lane);
    v16h b1 = load_b_frag(W, K, n0 + 16, k0, lane);
    acc00 = wmma_f16(a0, b0, acc00);
    acc01 = wmma_f16(a0, b1, acc01);
    acc10 = wmma_f16(a1, b0, acc10);
    acc11 = wmma_f16(a1, b1, acc11);
  }
  int c16 = lane & 15;
  int hi8 = (lane >> 4) << 3;
  const v8f* accs[4] = { &acc00, &acc01, &acc10, &acc11 };
#pragma unroll
  for (int t = 0; t < 4; ++t) {
    int i = t >> 1, j = t & 1;               // (M-subtile, N-subtile)
    int col = n0 + j * 16 + c16;
    float bv = bias[col];
    const v8f& a = *accs[(i << 1) | j];
    if (OUT16T) {
      v8h pk;
#pragma unroll
      for (int r = 0; r < 8; ++r) pk[r] = (h16)(a[r] + bv);
      // rows m0+i*16+hi8 .. +7 are contiguous in the transposed layout
      *(v8h*)(out16t + (size_t)col * M + m0 + i * 16 + hi8) = pk;
    } else {
#pragma unroll
      for (int r = 0; r < 8; ++r) {
        int row = m0 + i * 16 + r + hi8;
        float v = a[r] + bv;
        if (GELU)    v = gelu_exact(v);
        if (HAS_RES) v += resid[(size_t)row * Nc + col];
        if (OUT32)   out32[(size_t)row * Nc + col] = v;
        if (OUT16)   out16[(size_t)row * Nc + col] = (h16)v;
      }
    }
  }
}

// ------------------------- small dense matmul ------------------------------
// For the 128-row inducing-point path (negligible FLOPs). RESMODE:
// 0=none, 1=res[row][col], 2=res[row%16][col] (broadcast inds).
template <bool GELU, int RESMODE>
__global__ __launch_bounds__(256) void small_mm_kernel(
    const h16* __restrict__ A, const h16* __restrict__ W,
    const float* __restrict__ bias, const float* __restrict__ resid,
    float* __restrict__ out32, h16* __restrict__ out16,
    int M, int Nc, int K) {
  int idx = blockIdx.x * blockDim.x + threadIdx.x;
  if (idx >= M * Nc) return;
  int row = idx / Nc, col = idx - row * Nc;
  const h16* a = A + (size_t)row * K;
  const h16* w = W + (size_t)col * K;
  float acc = 0.f;
  for (int k = 0; k < K; ++k) acc += (float)a[k] * (float)w[k];
  acc += bias[col];
  if (GELU)         acc = gelu_exact(acc);
  if (RESMODE == 1) acc += resid[(size_t)row * Nc + col];
  if (RESMODE == 2) acc += resid[(size_t)(row & 15) * Nc + col];
  if (out32) out32[(size_t)row * Nc + col] = acc;
  if (out16) out16[(size_t)row * Nc + col] = (h16)acc;
}

__global__ void f32_to_f16_kernel(const float* __restrict__ s,
                                  h16* __restrict__ d, int n) {
  for (int i = blockIdx.x * blockDim.x + threadIdx.x; i < n;
       i += gridDim.x * blockDim.x)
    d[i] = (h16)s[i];
}

// ------------------- attention 1: 16 queries x 8192 keys -------------------
// One wave per (b,h); flash-style online softmax over key chunks of 32.
// V is consumed in per-head transposed layout Vt[d][token] so each P·V
// B-fragment is a single aligned 32B v16h load per lane.
__global__ __launch_bounds__(256) void attn1_kernel(
    const h16* __restrict__ Q1,   // [16 x 256] (shared across batch)
    const h16* __restrict__ K1,   // [B*SEQ x 256] row-major
    const h16* __restrict__ Vt,   // [256 x B*SEQ] transposed
    h16* __restrict__ out) {      // [B*16 x 256]
  __shared__ h16 p_lds[8][16 * 32];
  int lane = threadIdx.x & 31;
  int wave = threadIdx.x >> 5;
  int bh = blockIdx.x * 8 + wave;          // 0..63
  int b = bh >> 3, h = bh & 7;
  int col  = lane & 15;
  int kb16 = (lane >> 4) * 16;

  v16h qa = load_a_frag(Q1, DMODEL, 0, h * HEADD, lane);
  const h16* Kb  = K1 + (size_t)b * SEQ * DMODEL;
  const h16* Vt0 = Vt + (size_t)(h * HEADD + col) * BN + b * SEQ;       // hd=col
  const h16* Vt1 = Vt + (size_t)(h * HEADD + 16 + col) * BN + b * SEQ;  // hd=16+col

  float m_run[8], l_run[8];
  v8f o0 = {}, o1 = {};
#pragma unroll
  for (int r = 0; r < 8; ++r) { m_run[r] = -1e30f; l_run[r] = 0.f; }
  h16* pl = p_lds[wave];

  for (int t = 0; t < SEQ; t += 32) {
    v16h kb0 = load_b_frag(Kb, DMODEL, t,      h * HEADD, lane);
    v16h kb1 = load_b_frag(Kb, DMODEL, t + 16, h * HEADD, lane);
    v8f s0 = {}, s1 = {};
    s0 = wmma_f16(qa, kb0, s0);
    s1 = wmma_f16(qa, kb1, s1);
#pragma unroll
    for (int r = 0; r < 8; ++r) {
      float a0 = s0[r] * SM_SCALE, a1 = s1[r] * SM_SCALE;
      float mx = fmaxf(a0, a1);
#pragma unroll
      for (int mk = 8; mk >= 1; mk >>= 1) mx = fmaxf(mx, __shfl_xor(mx, mk, 16));
      float mnew = fmaxf(m_run[r], mx);
      float corr = __expf(m_run[r] - mnew);
      float e0 = __expf(a0 - mnew), e1 = __expf(a1 - mnew);
      float rs = e0 + e1;
#pragma unroll
      for (int mk = 8; mk >= 1; mk >>= 1) rs += __shfl_xor(rs, mk, 16);
      l_run[r] = l_run[r] * corr + rs;
      m_run[r] = mnew;
      o0[r] *= corr; o1[r] *= corr;
      int rowm = r + ((lane >> 4) << 3);
      pl[rowm * 32 + col]      = (h16)e0;
      pl[rowm * 32 + 16 + col] = (h16)e1;
    }
    // Re-fragment P (16x32) from LDS as WMMA A operand (DS is in-order per wave).
    v16h pa;
    {
      const h16* pp = pl + (lane & 15) * 32 + ((lane >> 4) * 16);
#pragma unroll
      for (int i = 0; i < 8; ++i) { pa[i] = pp[i]; pa[i + 8] = pp[i + 8]; }
    }
    // V fragments from transposed V: keys contiguous -> single v16h load.
    v16h vb0 = *(const v16h*)(Vt0 + t + kb16);
    v16h vb1 = *(const v16h*)(Vt1 + t + kb16);
    o0 = wmma_f16(pa, vb0, o0);
    o1 = wmma_f16(pa, vb1, o1);
  }
#pragma unroll
  for (int r = 0; r < 8; ++r) {
    int rowm = r + ((lane >> 4) << 3);
    float inv = 1.f / l_run[r];
    h16* orow = out + ((size_t)b * MIND + rowm) * DMODEL + h * HEADD;
    orow[col]      = (h16)(o0[r] * inv);
    orow[16 + col] = (h16)(o1[r] * inv);
  }
}

// ------------------- attention 2: 8192 queries x 16 keys -------------------
// One wave per (query-tile-of-16, head). Keys padded 16->32 with zeros.
__global__ __launch_bounds__(256) void attn2_kernel(
    const h16* __restrict__ Q2,   // [BN x 256]
    const h16* __restrict__ K2,   // [B*16 x 256]
    const h16* __restrict__ V2,   // [B*16 x 256]
    h16* __restrict__ out) {      // [BN x 256]
  __shared__ h16 p_lds[8][16 * 32];
  int lane = threadIdx.x & 31;
  int h = threadIdx.x >> 5;
  int tok0 = blockIdx.x * 16;
  int b = tok0 >> 13;                       // / SEQ
  int col  = lane & 15;
  int kb16 = (lane >> 4) * 16;

  v16h qa  = load_a_frag(Q2, DMODEL, tok0, h * HEADD, lane);
  v16h kbf = load_b_frag(K2, DMODEL, b * MIND, h * HEADD, lane);
  v8f s = {};
  s = wmma_f16(qa, kbf, s);

  h16* pl = p_lds[h];
#pragma unroll
  for (int r = 0; r < 8; ++r) {
    float a0 = s[r] * SM_SCALE;
    float mx = a0;
#pragma unroll
    for (int mk = 8; mk >= 1; mk >>= 1) mx = fmaxf(mx, __shfl_xor(mx, mk, 16));
    float e = __expf(a0 - mx);
    float rs = e;
#pragma unroll
    for (int mk = 8; mk >= 1; mk >>= 1) rs += __shfl_xor(rs, mk, 16);
    int rowm = r + ((lane >> 4) << 3);
    pl[rowm * 32 + col]      = (h16)(e / rs);
    pl[rowm * 32 + 16 + col] = (h16)0.f;    // zero-pad keys 16..31
  }
  v16h pa;
  {
    const h16* pp = pl + (lane & 15) * 32 + ((lane >> 4) * 16);
#pragma unroll
    for (int i = 0; i < 8; ++i) { pa[i] = pp[i]; pa[i + 8] = pp[i + 8]; }
  }
  v16h vb0, vb1;
#pragma unroll
  for (int i = 0; i < 16; ++i) {
    if (kb16 == 0) {
      const h16* vrow = V2 + ((size_t)(b * MIND + i)) * DMODEL + h * HEADD;
      vb0[i] = vrow[col]; vb1[i] = vrow[16 + col];
    } else {
      vb0[i] = (h16)0.f; vb1[i] = (h16)0.f;
    }
  }
  v8f o0 = {}, o1 = {};
  o0 = wmma_f16(pa, vb0, o0);
  o1 = wmma_f16(pa, vb1, o1);
#pragma unroll
  for (int r = 0; r < 8; ++r) {
    int rowm = r + ((lane >> 4) << 3);
    h16* orow = out + (size_t)(tok0 + rowm) * DMODEL + h * HEADD;
    orow[col]      = (h16)o0[r];
    orow[16 + col] = (h16)o1[r];
  }
}

// ------------------------------- launcher ----------------------------------
extern "C" void kernel_launch(void* const* d_in, const int* in_sizes, int n_in,
                              void* d_out, int out_size, void* d_ws, size_t ws_size,
                              hipStream_t stream) {
  // Input order: x, inds, then p1{in_w,in_b,out_w,out_b,ln1_w,ln1_b,ln2_w,
  // ln2_b,lin1_w,lin1_b,lin2_w,lin2_b}, then p2{same}.
  const float* x    = (const float*)d_in[0];
  const float* inds = (const float*)d_in[1];
  const float* p1[12]; const float* p2[12];
  for (int i = 0; i < 12; ++i) { p1[i] = (const float*)d_in[2 + i];
                                 p2[i] = (const float*)d_in[14 + i]; }
  enum { IN_W, IN_B, OUT_W, OUT_B, LN1_W, LN1_B, LN2_W, LN2_B,
         LIN1_W, LIN1_B, LIN2_W, LIN2_B };
  float* outp = (float*)d_out;

  // Workspace arena.
  size_t off = 0;
  auto carve = [&](size_t bytes) {
    void* p = (char*)d_ws + off;
    off += (bytes + 255) & ~(size_t)255;
    return p;
  };
  h16* in_w1h  = (h16*)carve((size_t)3 * 256 * 256 * 2);
  h16* out_w1h = (h16*)carve((size_t)256 * 256 * 2);
  h16* l1_w1h  = (h16*)carve((size_t)DFF * 256 * 2);
  h16* l2_w1h  = (h16*)carve((size_t)256 * DFF * 2);
  h16* in_w2h  = (h16*)carve((size_t)3 * 256 * 256 * 2);
  h16* out_w2h = (h16*)carve((size_t)256 * 256 * 2);
  h16* l1_w2h  = (h16*)carve((size_t)DFF * 256 * 2);
  h16* l2_w2h  = (h16*)carve((size_t)256 * DFF * 2);
  h16* xn_big  = (h16*)carve((size_t)BN * DMODEL * 2);  // xn1 / xn2 / h2n
  h16* k1_q2   = (h16*)carve((size_t)BN * DMODEL * 2);  // K1, later Q2
  h16* v1_ao2  = (h16*)carve((size_t)BN * DMODEL * 2);  // V1^T, later attn2 out
  h16* ffmid   = (h16*)carve((size_t)BN * DFF * 2);
  float* x2    = (float*)carve((size_t)BN * DMODEL * 4);
  h16* q1n     = (h16*)carve((size_t)MIND * DMODEL * 2);
  h16* Q1      = (h16*)carve((size_t)MIND * DMODEL * 2);
  h16* a1out   = (h16*)carve((size_t)BM * DMODEL * 2);
  float* x1    = (float*)carve((size_t)BM * DMODEL * 4);
  h16* h1n     = (h16*)carve((size_t)BM * DMODEL * 2);
  h16* ff1     = (h16*)carve((size_t)BM * DFF * 2);
  float* hbuf  = (float*)carve((size_t)BM * DMODEL * 4);
  h16* hn      = (h16*)carve((size_t)BM * DMODEL * 2);
  h16* K2      = (h16*)carve((size_t)BM * DMODEL * 2);
  h16* V2      = (h16*)carve((size_t)BM * DMODEL * 2);
  (void)ws_size; (void)in_sizes; (void)n_in; (void)out_size;

  auto cvt = [&](const float* s, h16* d, int n) {
    f32_to_f16_kernel<<<(n + 1023) / 1024, 256, 0, stream>>>(s, d, n);
  };
  cvt(p1[IN_W], in_w1h, 3 * 256 * 256);  cvt(p1[OUT_W], out_w1h, 256 * 256);
  cvt(p1[LIN1_W], l1_w1h, DFF * 256);    cvt(p1[LIN2_W], l2_w1h, 256 * DFF);
  cvt(p2[IN_W], in_w2h, 3 * 256 * 256);  cvt(p2[OUT_W], out_w2h, 256 * 256);
  cvt(p2[LIN1_W], l1_w2h, DFF * 256);    cvt(p2[LIN2_W], l2_w2h, 256 * DFF);

  dim3 blk(256);
  // ---------------- MAB1: h = MAB(inds, x, x; p1) ----------------
  layernorm_kernel<<<BN / 8, blk, 0, stream>>>(x, p1[LN1_W], p1[LN1_B], xn_big, BN);
  // K1 = xn1 · wk1ᵀ + bk1 (row-major) ; V1ᵀ = (xn1 · wv1ᵀ + bv1)ᵀ
  gemm_wmma_kernel<false, false, false, true, false>
      <<<dim3(BN / 128, 4), blk, 0, stream>>>(
      xn_big, in_w1h + 256 * 256, p1[IN_B] + 256, nullptr, nullptr, k1_q2, nullptr,
      BN, 256, 256);
  gemm_wmma_kernel<false, false, false, false, true>
      <<<dim3(BN / 128, 4), blk, 0, stream>>>(
      xn_big, in_w1h + 512 * 256, p1[IN_B] + 512, nullptr, nullptr, nullptr, v1_ao2,
      BN, 256, 256);
  // Q1 = ln(inds) · wq1ᵀ + bq1  (shared across batch)
  layernorm_kernel<<<2, blk, 0, stream>>>(inds, p1[LN1_W], p1[LN1_B], q1n, MIND);
  small_mm_kernel<false, 0><<<(MIND * 256 + 255) / 256, blk, 0, stream>>>(
      q1n, in_w1h, p1[IN_B], nullptr, nullptr, Q1, MIND, 256, 256);
  attn1_kernel<<<8, blk, 0, stream>>>(Q1, k1_q2, v1_ao2, a1out);
  // x1 = inds + attn1 · woᵀ + bo
  small_mm_kernel<false, 2><<<(BM * 256 + 255) / 256, blk, 0, stream>>>(
      a1out, out_w1h, p1[OUT_B], inds, x1, nullptr, BM, 256, 256);
  layernorm_kernel<<<BM / 8, blk, 0, stream>>>(x1, p1[LN2_W], p1[LN2_B], h1n, BM);
  small_mm_kernel<true, 0><<<(BM * DFF + 255) / 256, blk, 0, stream>>>(
      h1n, l1_w1h, p1[LIN1_B], nullptr, nullptr, ff1, BM, DFF, 256);
  small_mm_kernel<false, 1><<<(BM * 256 + 255) / 256, blk, 0, stream>>>(
      ff1, l2_w1h, p1[LIN2_B], x1, hbuf, nullptr, BM, 256, DFF);
  // ---------------- MAB2: out = MAB(x, h, h; p2) ----------------
  layernorm_kernel<<<BM / 8, blk, 0, stream>>>(hbuf, p2[LN1_W], p2[LN1_B], hn, BM);
  small_mm_kernel<false, 0><<<(BM * 256 + 255) / 256, blk, 0, stream>>>(
      hn, in_w2h + 256 * 256, p2[IN_B] + 256, nullptr, nullptr, K2, BM, 256, 256);
  small_mm_kernel<false, 0><<<(BM * 256 + 255) / 256, blk, 0, stream>>>(
      hn, in_w2h + 512 * 256, p2[IN_B] + 512, nullptr, nullptr, V2, BM, 256, 256);
  layernorm_kernel<<<BN / 8, blk, 0, stream>>>(x, p2[LN1_W], p2[LN1_B], xn_big, BN);
  gemm_wmma_kernel<false, false, false, true, false>
      <<<dim3(BN / 128, 4), blk, 0, stream>>>(
      xn_big, in_w2h, p2[IN_B], nullptr, nullptr, k1_q2, nullptr, BN, 256, 256); // Q2
  attn2_kernel<<<BN / 16, blk, 0, stream>>>(k1_q2, K2, V2, v1_ao2);
  // x2 = x + attn2 · wo2ᵀ + bo2
  gemm_wmma_kernel<false, true, true, false, false>
      <<<dim3(BN / 128, 4), blk, 0, stream>>>(
      v1_ao2, out_w2h, p2[OUT_B], x, x2, nullptr, nullptr, BN, 256, 256);
  layernorm_kernel<<<BN / 8, blk, 0, stream>>>(x2, p2[LN2_W], p2[LN2_B], xn_big, BN);
  // ffmid = gelu(h2n · lin1ᵀ + b1)
  gemm_wmma_kernel<true, false, false, true, false>
      <<<dim3(BN / 128, DFF / 64), blk, 0, stream>>>(
      xn_big, l1_w2h, p2[LIN1_B], nullptr, nullptr, ffmid, nullptr, BN, DFF, 256);
  // out = x2 + ffmid · lin2ᵀ + b2
  gemm_wmma_kernel<false, true, true, false, false>
      <<<dim3(BN / 128, 4), blk, 0, stream>>>(
      ffmid, l2_w2h, p2[LIN2_B], x2, outp, nullptr, nullptr, BN, 256, DFF);
}
